// TopKWindowAttention_27839978012822
// MI455X (gfx1250) — matrix operand order repaired and checked by reading, not compiled, not run on hardware
//
#include <hip/hip_runtime.h>
#include <hip/hip_bf16.h>
#include <stdint.h>

typedef __attribute__((ext_vector_type(16))) __bf16 v16bf;
typedef __attribute__((ext_vector_type(8)))  __bf16 v8bf;
typedef __attribute__((ext_vector_type(8)))  float  v8f;

#define BATCH 4
#define CHN 256
#define HPIX 112
#define WPIX 112
#define WS 7
#define NW 256                   // 16*16 windows per batch
#define LTOK 49                  // tokens per window
#define LPAD 64                  // padded tokens per window
#define TOPKN 8
#define NHEAD 8
#define DHEAD 32
#define PLANE  (HPIX*WPIX)       // 12544
#define NCHUNK 24                // 8 windows * 2 slices + 256 coarse / 32
#define EPSV   1e-6f

__device__ __forceinline__ float elu1(float x) {
  return x > 0.0f ? x + 1.0f : __expf(x);   // elu(x)+1
}

// merge two 8-element halves (each loaded as one b128) into a WMMA fragment
__device__ __forceinline__ v16bf frag2(const __bf16* p0, const __bf16* p1) {
  v8bf lo = *(const v8bf*)p0;
  v8bf hi = *(const v8bf*)p1;
  return __builtin_shufflevector(lo, hi, 0,1,2,3,4,5,6,7,8,9,10,11,12,13,14,15);
}

// ---------------------------------------------------------------------------
// Kernel A: per (b, channel) plane.
// mode 0 = q : token-major padded rows [b][w][64][c] (feat), + qm f32
// mode 1 = k : channel-major padded   [b][w][c][64] (feat), + km f32 + coarse feat-mean [b][c][w]
// mode 2 = v : channel-major padded   [b][w][c][64] (raw),  + coarse raw mean  [b][c][w]
// ---------------------------------------------------------------------------
__global__ __launch_bounds__(256)
void ka_featurize(const float* __restrict__ in, __bf16* __restrict__ fine,
                  float* __restrict__ meanOut, __bf16* __restrict__ coarse,
                  int mode)
{
  __shared__ float plane[PLANE];            // 49 KB
  const int b = blockIdx.x >> 8;
  const int c = blockIdx.x & 255;
  const float* src = in + ((size_t)(b * CHN + c)) * PLANE;
  for (int i = threadIdx.x; i < PLANE; i += 256) plane[i] = src[i];
  __syncthreads();

  if (mode == 0) {
    // token-major padded: qf[((b*NW+w)*LPAD + t)*CHN + c]
    for (int wt = threadIdx.x; wt < NW * LPAD; wt += 256) {
      int w = wt >> 6, t = wt & 63;
      float f = 0.f;
      if (t < LTOK) {
        int pr = (w >> 4) * WS + t / WS;
        int pc = (w & 15) * WS + t % WS;
        f = elu1(plane[pr * WPIX + pc]);
      }
      fine[((size_t)(b * NW + w) * LPAD + t) * CHN + c] = (__bf16)f;
    }
  } else {
    // channel-major padded: fineT[((b*NW+w)*CHN + c)*LPAD + t], 8-token packs
    for (int u = threadIdx.x; u < NW * 8; u += 256) {
      int w = u >> 3, tg = (u & 7) * 8;
      __bf16 tmp[8];
#pragma unroll
      for (int t2 = 0; t2 < 8; ++t2) {
        int t = tg + t2;
        float f = 0.f;
        if (t < LTOK) {
          int pr = (w >> 4) * WS + t / WS;
          int pc = (w & 15) * WS + t % WS;
          float x = plane[pr * WPIX + pc];
          f = (mode == 2) ? x : elu1(x);
        }
        tmp[t2] = (__bf16)f;
      }
      *(uint4*)(fine + ((size_t)(b * NW + w) * CHN + c) * LPAD + tg) = *(const uint4*)tmp;
    }
  }

  // per-window mean for this channel (one window per thread)
  int w = threadIdx.x;
  int pr0 = (w >> 4) * WS, pc0 = (w & 15) * WS;
  float s = 0.f;
  for (int t = 0; t < LTOK; ++t)
    s += plane[(pr0 + t / WS) * WPIX + (pc0 + t % WS)];
  float mean = s * (1.0f / LTOK);
  if (meanOut) meanOut[((size_t)b * NW + w) * CHN + c] = mean;
  if (coarse) {
    float f = (mode == 2) ? mean : elu1(mean);
    coarse[((size_t)b * CHN + c) * NW + w] = (__bf16)f;   // [b][c][w]
  }
}

// ---------------------------------------------------------------------------
// Kernel B: sim = qm @ km^T per (b, query window) row, stable top-8 argmax.
// ---------------------------------------------------------------------------
__global__ __launch_bounds__(256)
void kb_topk(const float* __restrict__ qm, const float* __restrict__ km,
             int* __restrict__ idxOut)
{
  __shared__ float s_qm[CHN];
  __shared__ float s_sim[NW];
  __shared__ float rv[NW];
  __shared__ int   ri[NW];
  const int b = blockIdx.x >> 8, qw = blockIdx.x & 255;
  const int tid = threadIdx.x;

  s_qm[tid] = qm[((size_t)b * NW + qw) * CHN + tid];
  __syncthreads();
  const float* kr = km + ((size_t)b * NW + tid) * CHN;
  float dot = 0.f;
  for (int c2 = 0; c2 < CHN; ++c2) dot += s_qm[c2] * kr[c2];
  s_sim[tid] = dot;
  __syncthreads();

  for (int k = 0; k < TOPKN; ++k) {
    rv[tid] = s_sim[tid]; ri[tid] = tid;
    __syncthreads();
    for (int off = 128; off > 0; off >>= 1) {
      if (tid < off) {
        float v1 = rv[tid + off]; int i1 = ri[tid + off];
        if (v1 > rv[tid] || (v1 == rv[tid] && i1 < ri[tid])) { rv[tid] = v1; ri[tid] = i1; }
      }
      __syncthreads();
    }
    if (tid == 0) { idxOut[(size_t)blockIdx.x * TOPKN + k] = ri[0]; s_sim[ri[0]] = -3.4e38f; }
    __syncthreads();
  }
}

// ---------------------------------------------------------------------------
// Kernel C: per query window linear attention, bf16 WMMA, wave h == head h.
// All fragment gathers are contiguous b128 LDS/global loads.
// ---------------------------------------------------------------------------
__global__ __launch_bounds__(256)
void kc_attn(const __bf16* __restrict__ qf,
             const __bf16* __restrict__ kfT, const __bf16* __restrict__ kcT,
             const __bf16* __restrict__ vfT, const __bf16* __restrict__ vcT,
             const int* __restrict__ topIdx, float* __restrict__ out)
{
  __shared__ __bf16 s_kT[CHN * 32];            // 16 KB [ch][32 tok] Kf^T slice
  __shared__ __bf16 s_vT[CHN * 32];            // 16 KB [ch][32 tok] V^T slice
  __shared__ __bf16 s_kvhT[NHEAD * 32 * 32];   // 16 KB KV^T per head [N][M]
  __shared__ __bf16 s_ksumh[NHEAD * 32];       // 512 B
  __shared__ int    s_widx[TOPKN];

  const int b  = blockIdx.x >> 8, qw = blockIdx.x & 255;
  const int tid = threadIdx.x;
  const int lane = tid & 31, h = tid >> 5;
  const int half = lane >> 4, lm = lane & 15;

  if (tid < TOPKN) s_widx[tid] = topIdx[(size_t)blockIdx.x * TOPKN + tid];
  __syncthreads();

  v8f zro = {0.f,0.f,0.f,0.f,0.f,0.f,0.f,0.f};
  v8f accKV00 = zro, accKV01 = zro, accKV10 = zro, accKV11 = zro;
  v8f accKs0 = zro, accKs1 = zro;
  v16bf bOnes;
#pragma unroll
  for (int i = 0; i < 16; ++i) bOnes[i] = (__bf16)1.0f;

  const int d0 = h * DHEAD + lm;

  for (int ch = 0; ch < NCHUNK; ++ch) {
    // slice source: 16 fine slices (8 windows x 2), then 8 coarse slices
    const __bf16* baseK; const __bf16* baseV; int strideT;
    if (ch < 16) {
      int g = ch >> 1, sl = ch & 1;
      size_t wb = (size_t)(b * NW + s_widx[g]) * CHN * LPAD + sl * 32;
      baseK = kfT + wb; baseV = vfT + wb; strideT = LPAD;
    } else {
      size_t cb = (size_t)b * CHN * NW + (ch - 16) * 32;
      baseK = kcT + cb; baseV = vcT + cb; strideT = NW;
    }
    __syncthreads();
    // stage [256 ch][32 tok] of Kf^T and V^T: 2048 b128 units
    for (int u = tid; u < 2048; u += 256) {
      int tsel = u >> 10; int uu = u & 1023;
      int row = uu >> 2, seg = (uu & 3) * 8;
      const __bf16* src = (tsel ? baseV : baseK) + (size_t)row * strideT + seg;
      uint4 val = *(const uint4*)src;
      *(uint4*)((tsel ? s_vT : s_kT) + row * 32 + seg) = val;
    }
    __syncthreads();

    // A = Kf^T (M = head dim, K = tok); two contiguous 8-K runs per lane
    v16bf a0 = frag2(s_kT + d0        * 32 + half * 8, s_kT + d0        * 32 + 16 + half * 8);
    v16bf a1 = frag2(s_kT + (d0 + 16) * 32 + half * 8, s_kT + (d0 + 16) * 32 + 16 + half * 8);
    // B = V (K = tok, N = head dim); two contiguous 8-K runs per lane
    v16bf b0 = frag2(s_vT + d0        * 32 + half * 8, s_vT + d0        * 32 + 16 + half * 8);
    v16bf b1 = frag2(s_vT + (d0 + 16) * 32 + half * 8, s_vT + (d0 + 16) * 32 + 16 + half * 8);

    accKV00 = __builtin_amdgcn_wmma_f32_16x16x32_bf16(false, a0, false, b0, (short)0, accKV00, false, false);
    accKV01 = __builtin_amdgcn_wmma_f32_16x16x32_bf16(false, a0, false, b1, (short)0, accKV01, false, false);
    accKV10 = __builtin_amdgcn_wmma_f32_16x16x32_bf16(false, a1, false, b0, (short)0, accKV10, false, false);
    accKV11 = __builtin_amdgcn_wmma_f32_16x16x32_bf16(false, a1, false, b1, (short)0, accKV11, false, false);
    accKs0  = __builtin_amdgcn_wmma_f32_16x16x32_bf16(false, a0, false, bOnes, (short)0, accKs0, false, false);
    accKs1  = __builtin_amdgcn_wmma_f32_16x16x32_bf16(false, a1, false, bOnes, (short)0, accKs1, false, false);
  }

  // dump KV transposed (bf16) so phase-2 B frags are K-contiguous.
  // D layout: elem j -> M = j + 8*half, N = lm.  s_kvhT[(h*32 + N)*32 + M]
#pragma unroll
  for (int j = 0; j < 8; ++j) {
    int m0 = j + half * 8;
    s_kvhT[(h * 32 + lm     ) * 32 + m0     ] = (__bf16)accKV00[j];
    s_kvhT[(h * 32 + 16 + lm) * 32 + m0     ] = (__bf16)accKV01[j];
    s_kvhT[(h * 32 + lm     ) * 32 + m0 + 16] = (__bf16)accKV10[j];
    s_kvhT[(h * 32 + 16 + lm) * 32 + m0 + 16] = (__bf16)accKV11[j];
  }
  if (lm == 0) {   // all N columns equal -> every lane holds Ksum
#pragma unroll
    for (int j = 0; j < 8; ++j) {
      s_ksumh[h * 32 + j + half * 8]      = (__bf16)accKs0[j];
      s_ksumh[h * 32 + 16 + j + half * 8] = (__bf16)accKs1[j];
    }
  }
  __syncthreads();

  // phase 2: msg = Qf @ KV ; denom via B = broadcast(Ksum)
  v16bf bkv0 = frag2(s_kvhT + (h * 32 + lm     ) * 32 + half * 16,
                     s_kvhT + (h * 32 + lm     ) * 32 + half * 16 + 8);
  v16bf bkv1 = frag2(s_kvhT + (h * 32 + 16 + lm) * 32 + half * 16,
                     s_kvhT + (h * 32 + 16 + lm) * 32 + half * 16 + 8);
  v16bf bden = frag2(s_ksumh + h * 32 + half * 16,
                     s_ksumh + h * 32 + half * 16 + 8);

  const __bf16* qbase = qf + (size_t)(b * NW + qw) * LPAD * CHN;
#pragma unroll
  for (int mt = 0; mt < 4; ++mt) {
    int tokA = mt * 16 + lm;                 // A-frag M = query token (zero-padded rows)
    const __bf16* qr = qbase + (size_t)tokA * CHN + h * DHEAD;
    v16bf aq = frag2(qr + half * 16, qr + half * 16 + 8);

    v8f zz = {0.f,0.f,0.f,0.f,0.f,0.f,0.f,0.f};
    v8f m0 = __builtin_amdgcn_wmma_f32_16x16x32_bf16(false, aq, false, bkv0, (short)0, zz, false, false);
    v8f m1 = __builtin_amdgcn_wmma_f32_16x16x32_bf16(false, aq, false, bkv1, (short)0, zz, false, false);
    v8f dn = __builtin_amdgcn_wmma_f32_16x16x32_bf16(false, aq, false, bden, (short)0, zz, false, false);
#pragma unroll
    for (int j = 0; j < 8; ++j) {
      int token = mt * 16 + j + half * 8;
      if (token < LTOK) {
        float z  = 1.0f / (dn[j] + EPSV);
        int pr = (qw >> 4) * WS + token / WS;
        int pc = (qw & 15) * WS + token % WS;
        size_t base = ((size_t)b * CHN) * PLANE + (size_t)pr * WPIX + pc;
        out[base + (size_t)(h * DHEAD + lm) * PLANE]      = m0[j] * z;
        out[base + (size_t)(h * DHEAD + 16 + lm) * PLANE] = m1[j] * z;
      }
    }
  }
}

// ---------------------------------------------------------------------------
extern "C" void kernel_launch(void* const* d_in, const int* in_sizes, int n_in,
                              void* d_out, int out_size, void* d_ws, size_t ws_size,
                              hipStream_t stream) {
  (void)in_sizes; (void)n_in; (void)out_size; (void)ws_size;
  const float* q = (const float*)d_in[0];
  const float* k = (const float*)d_in[1];
  const float* v = (const float*)d_in[2];
  float* out = (float*)d_out;

  char* ws = (char*)d_ws;
  size_t off = 0;
  auto alloc = [&](size_t bytes) -> void* {
    void* p = ws + off;
    off += (bytes + 255) & ~(size_t)255;
    return p;
  };
  __bf16* qfp = (__bf16*)alloc((size_t)BATCH * NW * LPAD * CHN * 2);  // token-major padded
  __bf16* kfT = (__bf16*)alloc((size_t)BATCH * NW * CHN * LPAD * 2);  // channel-major padded
  __bf16* vfT = (__bf16*)alloc((size_t)BATCH * NW * CHN * LPAD * 2);
  __bf16* kcT = (__bf16*)alloc((size_t)BATCH * CHN * NW * 2);         // coarse [b][c][w]
  __bf16* vcT = (__bf16*)alloc((size_t)BATCH * CHN * NW * 2);
  float*  qm  = (float*)alloc((size_t)BATCH * NW * CHN * 4);
  float*  km  = (float*)alloc((size_t)BATCH * NW * CHN * 4);
  int*    idx = (int*)alloc((size_t)BATCH * NW * TOPKN * 4);

  dim3 blk(256);
  ka_featurize<<<dim3(BATCH * CHN), blk, 0, stream>>>(q, qfp, qm, nullptr, 0);
  ka_featurize<<<dim3(BATCH * CHN), blk, 0, stream>>>(k, kfT, km, kcT, 1);
  ka_featurize<<<dim3(BATCH * CHN), blk, 0, stream>>>(v, vfT, nullptr, vcT, 2);
  kb_topk<<<dim3(BATCH * NW), blk, 0, stream>>>(qm, km, idx);
  kc_attn<<<dim3(BATCH * NW), blk, 0, stream>>>(qfp, kfT, kcT, vfT, vcT, idx, out);
}
